// PPOSurpriseGRUPolicy_8263517077708
// MI455X (gfx1250) — matrix operand order repaired and checked by reading, not compiled, not run on hardware
//
#include <hip/hip_runtime.h>
#include <hip/hip_bf16.h>
#include <stdint.h>

typedef __bf16 bf16;
typedef __attribute__((ext_vector_type(16))) __bf16 v16bf;
typedef __attribute__((ext_vector_type(8)))  float  v8f;
typedef __attribute__((ext_vector_type(4)))  unsigned int v4u;
typedef __attribute__((ext_vector_type(8)))  int v8i;
typedef __attribute__((ext_vector_type(4)))  int v4i;

#define NTHREADS 256
#define NWAVES 8

__device__ __forceinline__ bf16 to_bf16(float f) {
  union { float f; uint32_t u; } v; v.f = f;
  uint32_t r = v.u + 0x7FFFu + ((v.u >> 16) & 1u);   // round-to-nearest-even
  uint16_t h = (uint16_t)(r >> 16);
  union { bf16 b; uint16_t s; } o; o.s = h;
  return o.b;
}

__device__ __forceinline__ float sigmoidf_(float x) { return 1.0f / (1.0f + __expf(-x)); }
__device__ __forceinline__ float leakyf_(float x)   { return x > 0.f ? x : 0.01f * x; }
__device__ __forceinline__ float softplusf_(float x){ return x > 20.f ? x : log1pf(__expf(x)); }

// One 16x16 D tile of A(16xK) * B(KxN), B given transposed (Bt: N x K row-major).
// A-fragment per ISA 16-bit 16x32 layout; B-fragment uses the same K order as A.
__device__ __forceinline__ v8f wmma_km(const bf16* __restrict__ A, int lda,
                                       const bf16* __restrict__ Bt, int ldb,
                                       int K, int col0, v8f acc)
{
  const int lane = threadIdx.x & 31;
  const int half = lane >> 4;
  const int r    = lane & 15;
  const uint32_t* A32 = (const uint32_t*)(A + (size_t)r * lda);
  const uint32_t* B32 = (const uint32_t*)(Bt + (size_t)(col0 + r) * ldb);
  for (int kb = 0; kb < K; kb += 32) {
    union { v16bf v; uint32_t u[8]; } a, b;
    const int kd = kb >> 1;
#pragma unroll
    for (int j = 0; j < 4; ++j) a.u[j]     = A32[kd + half * 4 + j];        // K = half*8 + 2j (+1)
#pragma unroll
    for (int j = 0; j < 4; ++j) a.u[4 + j] = A32[kd + 8 + half * 4 + j];    // K = 16 + half*8 + 2j
#pragma unroll
    for (int j = 0; j < 4; ++j) b.u[j]     = B32[kd + half * 4 + j];        // K = half*8 + 2j (+1)
#pragma unroll
    for (int j = 0; j < 4; ++j) b.u[4 + j] = B32[kd + 8 + half * 4 + j];    // K = 16 + half*8 + 2j
    acc = __builtin_amdgcn_wmma_f32_16x16x32_bf16(false, a.v, false, b.v,
                                                  (short)0, acc, false, false);
  }
  return acc;
}

// C/D layout: lanes 0-15 -> N=lane, VGPR rr -> M=rr; lanes 16-31 -> M=8+rr.
__device__ __forceinline__ void store_tile16(float* __restrict__ Out, int ldo, int col0, v8f c) {
  const int lane = threadIdx.x & 31;
  const int half = lane >> 4;
  const int n    = lane & 15;
#pragma unroll
  for (int rr = 0; rr < 8; ++rr)
    Out[(size_t)(half * 8 + rr) * ldo + col0 + n] = c[rr];
}

// 16 x N = (16 x K) @ (K x N), column tiles split over 8 waves.
__device__ __forceinline__ void gemm16(const bf16* A, int lda, const bf16* Bt, int ldb,
                                       int K, int N, float* Out, int ldo)
{
  const int wave = threadIdx.x >> 5;
  const int nt = N >> 4;
  for (int tn = wave; tn < nt; tn += NWAVES) {
    v8f acc = {};
    acc = wmma_km(A, lda, Bt, ldb, K, tn * 16, acc);
    store_tile16(Out, ldo, tn * 16, acc);
  }
}

// Split-K concat GEMM: D = [A0 | A1] @ B, each chunk Khalf wide. Bt is N x (2*Khalf).
__device__ __forceinline__ void gemm16_cat(const bf16* A0, const bf16* A1, int lda,
                                           const bf16* Bt, int ldb, int Khalf,
                                           int N, float* Out, int ldo)
{
  const int wave = threadIdx.x >> 5;
  const int nt = N >> 4;
  for (int tn = wave; tn < nt; tn += NWAVES) {
    v8f acc = {};
    acc = wmma_km(A0, lda, Bt,         ldb, Khalf, tn * 16, acc);
    acc = wmma_km(A1, lda, Bt + Khalf, ldb, Khalf, tn * 16, acc);
    store_tile16(Out, ldo, tn * 16, acc);
  }
}

// TDM: async 2D tile load (16 rows x 256 bf16, row stride 65536 elems) into LDS.
// D# group0/group1 packed per CDNA5 ISA ch.8 (count=1, type=2, data_size=2B).
// This toolchain's builtin is the 6-arg form: (g0, g1, g2, g3, g_extra, cpol).
__device__ __forceinline__ void tdm_load_x(uint32_t lds_off, uint64_t ga) {
  v4u g0;
  g0[0] = 1u;                                               // count=1, user mode
  g0[1] = lds_off;                                          // lds_addr[31:0]
  g0[2] = (uint32_t)ga;                                     // global_addr[31:0]
  g0[3] = (uint32_t)((ga >> 32) & 0x1FFFFFFu) | (2u << 30); // global_addr[56:32] | type=2
  v8i g1;
  g1[0] = (int)(1u << 16);       // workgroup_mask=0 | data_size=1 (2 bytes)
  g1[1] = (int)(256u << 16);     // atomic_addr=0 | tensor_dim0.lo16 = 256
  g1[2] = (int)(16u << 16);      // tensor_dim0.hi16=0 | tensor_dim1.lo16 = 16
  g1[3] = (int)(256u << 16);     // tensor_dim1.hi16=0 | tile_dim0 = 256
  g1[4] = (int)(16u);            // tile_dim1 = 16 | tile_dim2 = 0
  g1[5] = (int)(65536u);         // tensor_dim0_stride.lo32 = 65536 elems (T*E)
  g1[6] = 0;                     // stride0.hi16 | stride1.lo16
  g1[7] = 0;                     // stride1.hi32
  v4i z4 = {0, 0, 0, 0};
  v8i z8 = {0, 0, 0, 0, 0, 0, 0, 0};
  __builtin_amdgcn_tensor_load_to_lds(g0, g1, z4, z4, z8, 0);
}

// ---------- weight convert + transpose: W (KxN f32, row-major) -> Wt (NxK bf16) ----------
__global__ void wconv_kernel(const float* __restrict__ W, bf16* __restrict__ Wt, int K, int N) {
  int i = blockIdx.x * blockDim.x + threadIdx.x;
  if (i >= K * N) return;
  int k = i / N, n = i - k * N;
  Wt[(size_t)n * K + k] = to_bf16(W[i]);
}

// ---------- time-parallel encoder: x = leaky(leaky(LN(obs@obs_W+b)@enc_W+b)@pre_W+b) ----------
__global__ __launch_bounds__(NTHREADS) void precompute_kernel(
    const float* __restrict__ obs,
    const bf16* __restrict__ obs_Wt, const float* __restrict__ obs_b,
    const bf16* __restrict__ enc_Wt, const float* __restrict__ enc_b,
    const bf16* __restrict__ pre_Wt, const float* __restrict__ pre_b,
    bf16* __restrict__ x_ws)
{
  __shared__ bf16  s_obs[16 * 64];
  __shared__ float s_f[16 * 256];
  __shared__ bf16  s_e[16 * 256];
  __shared__ bf16  s_e2[16 * 256];
  const int tid = threadIdx.x;
  const size_t tok0 = (size_t)blockIdx.x * 16;

  for (int i = tid; i < 16 * 64; i += NTHREADS) {
    int r = i >> 6, c = i & 63;
    s_obs[i] = to_bf16(obs[(tok0 + r) * 64 + c]);
  }
  __syncthreads();
  gemm16(s_obs, 64, obs_Wt, 64, 64, 256, s_f, 256);
  __syncthreads();
  if (tid < 16) {                       // per-row layernorm over E=256
    int r = tid;
    float mu = 0.f;
    for (int c = 0; c < 256; ++c) mu += s_f[r * 256 + c] + obs_b[c];
    mu *= (1.f / 256.f);
    float var = 0.f;
    for (int c = 0; c < 256; ++c) { float v = s_f[r * 256 + c] + obs_b[c] - mu; var += v * v; }
    var *= (1.f / 256.f);
    float inv = rsqrtf(var + 1e-5f);
    for (int c = 0; c < 256; ++c)
      s_e[r * 256 + c] = to_bf16((s_f[r * 256 + c] + obs_b[c] - mu) * inv);
  }
  __syncthreads();
  gemm16(s_e, 256, enc_Wt, 256, 256, 256, s_f, 256);
  __syncthreads();
  for (int i = tid; i < 16 * 256; i += NTHREADS) {
    int c = i & 255;
    s_e2[i] = to_bf16(leakyf_(s_f[i] + enc_b[c]));
  }
  __syncthreads();
  gemm16(s_e2, 256, pre_Wt, 256, 256, 256, s_f, 256);
  __syncthreads();
  for (int i = tid; i < 16 * 256; i += NTHREADS) {
    int r = i >> 8, c = i & 255;
    x_ws[(tok0 + r) * 256 + c] = to_bf16(leakyf_(s_f[i] + pre_b[c]));
  }
}

// ---------- sequential scan: 64 blocks x 16 batch rows, h resident in LDS ----------
// x_t tiles are double-buffered via TDM (tensor_load_to_lds) one step ahead.
__global__ __launch_bounds__(NTHREADS) void scan_kernel(
    const float* __restrict__ h0, const int* __restrict__ epoch_p,
    const bf16* __restrict__ x_ws,
    const bf16* __restrict__ pred_Wt, const float* __restrict__ pred_b,
    const bf16* __restrict__ Wrz_Wt,  const float* __restrict__ Wrz_b,
    const bf16* __restrict__ Wn_Wt,   const float* __restrict__ Wn_b,
    const bf16* __restrict__ g1_Wt,   const float* __restrict__ g1_b,
    const float* __restrict__ ln_g,   const float* __restrict__ ln_b,
    const bf16* __restrict__ g2_Wt,   const float* __restrict__ g2_b,
    const bf16* __restrict__ g3_Wt,   const float* __restrict__ g3_b,
    const bf16* __restrict__ dp_Wt,
    float* __restrict__ out)
{
  extern __shared__ char smem[];
  bf16*  x0    = (bf16*) (smem + 0);       // 16x256 bf16 (TDM stage, ping)
  bf16*  x1    = (bf16*) (smem + 8192);    // 16x256 bf16 (TDM stage, pong)
  bf16*  h_bf  = (bf16*) (smem + 16384);   // 16x256
  bf16*  rh_bf = (bf16*) (smem + 24576);   // 16x256 (r*h)
  float* h_f   = (float*)(smem + 32768);   // 16x256
  float* d_f   = (float*)(smem + 49152);   // 16x256 delta
  bf16*  d_bf  = (bf16*) (smem + 65536);   // 16x256
  float* rz_f  = (float*)(smem + 73728);   // 16x512 (post-sigmoid)
  float* n_f   = (float*)(smem + 106496);  // 16x256
  float* dh_f  = (float*)(smem + 122880);  // 16x256
  float* g_f   = (float*)(smem + 139264);  // 16x32
  bf16*  g_bf  = (bf16*) (smem + 141312);  // 16x32
  bf16*  g2_bf = (bf16*) (smem + 142336);  // 16x32
  float* g3_f  = (float*)(smem + 143360);  // 16x768

  const int tid = threadIdx.x;
  const size_t rb0 = (size_t)blockIdx.x * 16;
  const int ep = epoch_p[0];
  const float scale = (ep < 5) ? 0.f : fminf(1.f, (float)(ep - 5) * (1.f / 40.f));

  // Wave-uniform scalar (forces a real s_cbranch around the EXEC-ignoring TDM op).
  const int waveid = __builtin_amdgcn_readfirstlane((int)(threadIdx.x >> 5));
  const uint32_t x0_off = (uint32_t)(size_t)x0;   // low 32 bits of generic ptr = LDS offset
  const uint32_t x1_off = (uint32_t)(size_t)x1;
  const uint64_t x_base = (uint64_t)(size_t)x_ws + (uint64_t)rb0 * 256u * 512u; // +t*512B

  // Kick off TDM for t=0 into x0.
  if (waveid == 0) tdm_load_x(x0_off, x_base);

  for (int i = tid; i < 16 * 256; i += NTHREADS) {
    int r = i >> 8, c = i & 255;
    float h = h0[(rb0 + r) * 256 + c];
    h_f[i] = h;
    h_bf[i] = to_bf16(h);
  }
  if (waveid == 0) __builtin_amdgcn_s_wait_tensorcnt(0);
  __syncthreads();

  for (int t = 0; t < 256; ++t) {
    const bf16* xa = (t & 1) ? x1 : x0;
    // Prefetch x for t+1 into the other stage (overlaps with this whole step).
    if (waveid == 0 && (t + 1) < 256)
      tdm_load_x((t & 1) ? x0_off : x1_off, x_base + (uint64_t)(t + 1) * 512u);

    gemm16(h_bf, 256, pred_Wt, 256, 256, 256, d_f, 256);              // h @ pred_W
    gemm16_cat(xa, h_bf, 256, Wrz_Wt, 512, 256, 512, rz_f, 512);      // [x,h] @ Wrz
    __syncthreads();

    for (int i = tid; i < 16 * 256; i += NTHREADS) {
      int c = i & 255;
      float d = d_f[i] + pred_b[c] - h_f[i];
      d_f[i] = d;
      d_bf[i] = to_bf16(d);
    }
    for (int i = tid; i < 16 * 512; i += NTHREADS) {
      int r = i >> 9, c = i & 511;
      float v = sigmoidf_(rz_f[i] + Wrz_b[c]);
      rz_f[i] = v;
      if (c < 256) rh_bf[r * 256 + c] = to_bf16(v * h_f[r * 256 + c]);  // r*h
    }
    __syncthreads();

    gemm16_cat(xa, rh_bf, 256, Wn_Wt, 512, 256, 256, n_f, 256);       // [x,r*h] @ Wn
    gemm16(d_bf, 256, g1_Wt, 256, 256,  32, g_f,  32);                // delta @ g1
    gemm16(d_bf, 256, dp_Wt, 256, 256, 256, dh_f, 256);               // delta @ dp
    __syncthreads();

    for (int i = tid; i < 16 * 256; i += NTHREADS) {
      int c = i & 255;
      n_f[i] = tanhf(n_f[i] + Wn_b[c]);
    }
    if (tid < 16) {                                                   // LN(G=32) + relu
      int r = tid;
      float mu = 0.f;
      for (int c = 0; c < 32; ++c) mu += g_f[r * 32 + c] + g1_b[c];
      mu *= (1.f / 32.f);
      float var = 0.f;
      for (int c = 0; c < 32; ++c) { float v = g_f[r * 32 + c] + g1_b[c] - mu; var += v * v; }
      var *= (1.f / 32.f);
      float inv = rsqrtf(var + 1e-5f);
      for (int c = 0; c < 32; ++c) {
        float v = (g_f[r * 32 + c] + g1_b[c] - mu) * inv * ln_g[c] + ln_b[c];
        g_bf[r * 32 + c] = to_bf16(fmaxf(v, 0.f));
      }
    }
    __syncthreads();

    gemm16(g_bf, 32, g2_Wt, 32, 32, 32, g_f, 32);
    __syncthreads();
    for (int i = tid; i < 16 * 32; i += NTHREADS) {
      int c = i & 31;
      g2_bf[i] = to_bf16(fmaxf(g_f[i] + g2_b[c], 0.f));
    }
    __syncthreads();
    gemm16(g2_bf, 32, g3_Wt, 32, 32, 768, g3_f, 768);
    __syncthreads();

    for (int i = tid; i < 16 * 256; i += NTHREADS) {                  // gate + update h
      int r = i >> 8, c = i & 255;
      float h  = h_f[i];
      float z  = rz_f[r * 512 + 256 + c];
      float n  = n_f[i];
      float eta = sigmoidf_(g3_f[r * 768 + c] + g3_b[c]);
      float th  = fminf(softplusf_(g3_f[r * 768 + 256 + c] + g3_b[256 + c]), 5.f);
      float al  = sigmoidf_(g3_f[r * 768 + 512 + c] + g3_b[512 + c]) * scale;
      float s   = fminf(fmaxf(eta * h - th * dh_f[i], -5.f), 5.f);
      float ht  = (1.f - al) * h + al * s;
      float hn  = z * ht + (1.f - z) * n;
      h_f[i] = hn;
      h_bf[i] = to_bf16(hn);
      out[((rb0 + r) * 256 + (size_t)t) * 256 + c] = hn;
    }
    // Ensure t+1's x tile has landed before next step reads it.
    if (waveid == 0) __builtin_amdgcn_s_wait_tensorcnt(0);
    __syncthreads();
  }
}

extern "C" void kernel_launch(void* const* d_in, const int* in_sizes, int n_in,
                              void* d_out, int out_size, void* d_ws, size_t ws_size,
                              hipStream_t stream)
{
  (void)in_sizes; (void)n_in; (void)out_size; (void)ws_size;
  const float* obs    = (const float*)d_in[0];
  const float* h0     = (const float*)d_in[1];
  const int*   epoch  = (const int*)  d_in[2];
  const float* obs_W  = (const float*)d_in[3];
  const float* obs_b  = (const float*)d_in[4];
  const float* enc_W  = (const float*)d_in[5];
  const float* enc_b  = (const float*)d_in[6];
  const float* pre_W  = (const float*)d_in[7];
  const float* pre_b  = (const float*)d_in[8];
  const float* pred_W = (const float*)d_in[9];
  const float* pred_b = (const float*)d_in[10];
  const float* Wrz_W  = (const float*)d_in[11];
  const float* Wrz_b  = (const float*)d_in[12];
  const float* Wn_W   = (const float*)d_in[13];
  const float* Wn_b   = (const float*)d_in[14];
  const float* g1_W   = (const float*)d_in[15];
  const float* g1_b   = (const float*)d_in[16];
  const float* ln_g   = (const float*)d_in[17];
  const float* ln_b   = (const float*)d_in[18];
  const float* g2_W   = (const float*)d_in[19];
  const float* g2_b   = (const float*)d_in[20];
  const float* g3_W   = (const float*)d_in[21];
  const float* g3_b   = (const float*)d_in[22];
  const float* dp_W   = (const float*)d_in[23];
  float* out = (float*)d_out;

  char* ws = (char*)d_ws;
  size_t off = 0;
  auto carve = [&](size_t bytes) -> void* {
    void* p = ws + off; off += (bytes + 255) & ~(size_t)255; return p;
  };
  const size_t BT = 1024ull * 256ull;
  bf16* x_ws    = (bf16*)carve(BT * 256 * 2);
  bf16* obs_Wt  = (bf16*)carve(256 * 64 * 2);
  bf16* enc_Wt  = (bf16*)carve(256 * 256 * 2);
  bf16* pre_Wt  = (bf16*)carve(256 * 256 * 2);
  bf16* pred_Wt = (bf16*)carve(256 * 256 * 2);
  bf16* Wrz_Wt  = (bf16*)carve(512 * 512 * 2);
  bf16* Wn_Wt   = (bf16*)carve(256 * 512 * 2);
  bf16* g1_Wt   = (bf16*)carve(32 * 256 * 2);
  bf16* g2_Wt   = (bf16*)carve(32 * 32 * 2);
  bf16* g3_Wt   = (bf16*)carve(768 * 32 * 2);
  bf16* dp_Wt   = (bf16*)carve(256 * 256 * 2);

  auto conv = [&](const float* W, bf16* Wt, int K, int N) {
    int total = K * N;
    wconv_kernel<<<(total + 255) / 256, 256, 0, stream>>>(W, Wt, K, N);
  };
  conv(obs_W,  obs_Wt,  64, 256);
  conv(enc_W,  enc_Wt,  256, 256);
  conv(pre_W,  pre_Wt,  256, 256);
  conv(pred_W, pred_Wt, 256, 256);
  conv(Wrz_W,  Wrz_Wt,  512, 512);
  conv(Wn_W,   Wn_Wt,   512, 256);
  conv(g1_W,   g1_Wt,   256, 32);
  conv(g2_W,   g2_Wt,   32, 32);
  conv(g3_W,   g3_Wt,   32, 768);
  conv(dp_W,   dp_Wt,   256, 256);

  precompute_kernel<<<(int)(BT / 16), NTHREADS, 0, stream>>>(
      obs, obs_Wt, obs_b, enc_Wt, enc_b, pre_Wt, pre_b, x_ws);

  const size_t lds_bytes = 143360 + 16 * 768 * 4;   // 192512 B < 320 KB/WGP
  scan_kernel<<<1024 / 16, NTHREADS, lds_bytes, stream>>>(
      h0, epoch, x_ws,
      pred_Wt, pred_b, Wrz_Wt, Wrz_b, Wn_Wt, Wn_b,
      g1_Wt, g1_b, ln_g, ln_b, g2_Wt, g2_b, g3_Wt, g3_b, dp_Wt, out);
}
